// GNN_predictor_17566416241198
// MI455X (gfx1250) — matrix operand order repaired and checked by reading, not compiled, not run on hardware
//
#include <hip/hip_runtime.h>

// ---------------------------------------------------------------------------
// GNN predictor for MI455X (gfx1250): bf16 WMMA pipeline, 2x2 register-blocked
//   prep:   A^T = relu(m) -> bf16 (padded 128x128), dinv = rsqrt(colsum+1)
//           x -> bf16 padded, W1/W2 -> bf16 transposed [n][k]
//   layer:  xsT = dinv * (X @ W)                    (WMMA bf16, f32 acc)
//           H   = lrelu(dinv*(A^T@xs + xs) + b)     (WMMA bf16, f32 acc)
//   final:  out = H.flat @ Wc^T + bc                (f32 block reduction)
// ---------------------------------------------------------------------------

typedef __bf16 bf16;
typedef __attribute__((ext_vector_type(16))) __bf16 v16bf;
typedef __attribute__((ext_vector_type(8)))  __bf16 v8bf;
typedef __attribute__((ext_vector_type(8)))  float  v8f;

#define BZ  512
#define R   116
#define DIN 116
#define H   256
#define RP  128   // R padded to 8 WMMA tiles
#define DP  128   // DIN padded

union FragBF { v16bf v; v8bf h[2]; };

// A-operand: 16x32 bf16 tile, row-major source, row stride = ldk.
// lanes 0-15: row m0+lane, K = k0+{0..7, 16..23}; lanes 16-31: K = k0+{8..15, 24..31}
__device__ inline v16bf load_a_frag(const bf16* __restrict__ A, int row, int k0,
                                    int ldk, int hs) {
  FragBF f;
  const bf16* p = A + (size_t)row * ldk + k0 + 8 * hs;
  f.h[0] = *(const v8bf*)p;
  f.h[1] = *(const v8bf*)(p + 16);
  return f.v;
}

// B-operand: 32x16 bf16 tile from K-contiguous (column-transposed) source Bt[n][k].
// lanes 0-15: col n0+lane, K = k0+{0..7, 16..23} ; lanes 16-31: col n0+lane-16, K = k0+{8..15, 24..31}
__device__ inline v16bf load_b_frag(const bf16* __restrict__ Bt, int col, int k0,
                                    int ldk, int hs) {
  FragBF f;
  const bf16* p = Bt + (size_t)col * ldk + k0 + 8 * hs;
  f.h[0] = *(const v8bf*)p;
  f.h[1] = *(const v8bf*)(p + 16);
  return f.v;
}

__device__ inline v8f wmma_bf16(v16bf a, v16bf b, v8f c) {
  return __builtin_amdgcn_wmma_f32_16x16x32_bf16(
      false, a, false, b, (short)0, c, false, false);
}

// --------------------------- prep kernels ----------------------------------

// One block per batch; thread = column c. deg[c] = sum_r relu(m[r,c]) + 1.
// Writes At[b][c][r] (A-transposed, bf16, zero-padded to 128x128), dinv[b][c].
__global__ __launch_bounds__(RP) void prep_adj_kernel(
    const float* __restrict__ m, bf16* __restrict__ At, float* __restrict__ dinv) {
  const int b = blockIdx.x;
  const int c = threadIdx.x;
  bf16* row = At + (size_t)b * RP * RP + (size_t)c * RP;
  if (c < R) {
    const float* mb = m + (size_t)b * R * R;
    float s = 0.f;
    for (int r = 0; r < R; ++r) {
      float v = mb[(size_t)r * R + c];          // coalesced across threads
      float a = v > 0.f ? v : 0.f;
      s += a;
      row[r] = (bf16)a;
    }
    for (int r = R; r < RP; ++r) row[r] = (bf16)0.f;
    dinv[(size_t)b * RP + c] = rsqrtf(s + 1.f);
  } else {
    for (int r = 0; r < RP; ++r) row[r] = (bf16)0.f;
    dinv[(size_t)b * RP + c] = 0.f;             // pad rows contribute nothing
  }
}

// One block per batch; thread = feature k. Xb[b][r][k] bf16, zero-padded.
__global__ __launch_bounds__(DP) void prep_x_kernel(
    const float* __restrict__ nf, bf16* __restrict__ Xb) {
  const int b = blockIdx.x;
  const int k = threadIdx.x;
  bf16* xb = Xb + (size_t)b * RP * DP;
  for (int r = 0; r < RP; ++r) {
    float v = (r < R && k < DIN) ? nf[((size_t)b * R + r) * DIN + k] : 0.f;
    xb[(size_t)r * DP + k] = (bf16)v;
  }
}

// W1t[n][k] = bf16(W1[k][n]) with K padded 116->128
__global__ __launch_bounds__(DP) void prep_w1_kernel(
    const float* __restrict__ W1, bf16* __restrict__ W1t) {
  const int n = blockIdx.x;   // 0..255
  const int k = threadIdx.x;  // 0..127
  W1t[(size_t)n * DP + k] = (bf16)((k < DIN) ? W1[(size_t)k * H + n] : 0.f);
}

// W2t[n][k] = bf16(W2[k][n]), 256x256
__global__ __launch_bounds__(H) void prep_w2_kernel(
    const float* __restrict__ W2, bf16* __restrict__ W2t) {
  const int n = blockIdx.x;
  const int k = threadIdx.x;
  W2t[(size_t)n * H + k] = (bf16)W2[(size_t)k * H + n];
}

// --------------------------- WMMA GEMM kernels -----------------------------
// Each wave computes a 32x32 output patch (2x2 tiles of 16x16), double-buffered
// over K. 8 waves/block; 32 wave-tiles per batch -> grid (4, BZ).

// xsT[b][f][r] = dinv[b][r] * sum_k A[b][r][k] * Wt[f][k]
// A: [BZ][RP x K] row-major (stride K); Wt: [H x K]; output transposed [H x RP].
template <int K>
__global__ __launch_bounds__(256) void gemm_xs_kernel(
    const bf16* __restrict__ A, const bf16* __restrict__ Wt,
    const float* __restrict__ dinv, bf16* __restrict__ xsT) {
  const int b    = blockIdx.y;
  const int wave = threadIdx.x >> 5;
  const int lane = threadIdx.x & 31;
  const int idx  = blockIdx.x * 8 + wave;       // 0..31
  const int m0 = (idx & 3) << 5;                // 4 M-groups of 32 (RP=128)
  const int n0 = (idx >> 2) << 5;               // 8 N-groups of 32 (H=256)
  const int hs = lane >> 4;
  const int ln = lane & 15;

  const bf16* Ab = A + (size_t)b * RP * K;

  v16bf a0 = load_a_frag(Ab, m0 + ln,      0, K, hs);
  v16bf a1 = load_a_frag(Ab, m0 + 16 + ln, 0, K, hs);
  v16bf b0 = load_b_frag(Wt, n0 + ln,      0, K, hs);
  v16bf b1 = load_b_frag(Wt, n0 + 16 + ln, 0, K, hs);
  v8f acc00 = {}, acc01 = {}, acc10 = {}, acc11 = {};

#pragma unroll
  for (int k0 = 0; k0 < K; k0 += 32) {
    v16bf na0 = a0, na1 = a1, nb0 = b0, nb1 = b1;
    if (k0 + 32 < K) {                          // prefetch next K-step (regs)
      na0 = load_a_frag(Ab, m0 + ln,      k0 + 32, K, hs);
      na1 = load_a_frag(Ab, m0 + 16 + ln, k0 + 32, K, hs);
      nb0 = load_b_frag(Wt, n0 + ln,      k0 + 32, K, hs);
      nb1 = load_b_frag(Wt, n0 + 16 + ln, k0 + 32, K, hs);
    }
    acc00 = wmma_bf16(a0, b0, acc00);
    acc01 = wmma_bf16(a0, b1, acc01);
    acc10 = wmma_bf16(a1, b0, acc10);
    acc11 = wmma_bf16(a1, b1, acc11);
    a0 = na0; a1 = na1; b0 = nb0; b1 = nb1;
  }

  // C layout: element i -> row m + i + 8*hs, col n + ln (rows contiguous in i)
  bf16* xsb = xsT + (size_t)b * (H * RP);
  const float* dvb = dinv + (size_t)b * RP;
  const v8f* accs[2][2] = {{&acc00, &acc01}, {&acc10, &acc11}};
#pragma unroll
  for (int i = 0; i < 2; ++i) {
    const int r0 = m0 + 16 * i + 8 * hs;
    const float* dv = dvb + r0;
#pragma unroll
    for (int j = 0; j < 2; ++j) {
      const int f = n0 + 16 * j + ln;
      const v8f& acc = *accs[i][j];
      v8bf outv;
#pragma unroll
      for (int e = 0; e < 8; ++e) outv[e] = (bf16)(dv[e] * acc[e]);
      *(v8bf*)(xsb + (size_t)f * RP + r0) = outv;
    }
  }
}

// Hout[b][c][f] = lrelu( dinv[c] * (sum_r At[b][c][r]*xsT[b][f][r] + xs[c][f]) + bias[f] )
__global__ __launch_bounds__(256) void gemm_agg_kernel(
    const bf16* __restrict__ At, const bf16* __restrict__ xsT,
    const float* __restrict__ dinv, const float* __restrict__ bias,
    bf16* __restrict__ Hout) {
  const int b    = blockIdx.y;
  const int wave = threadIdx.x >> 5;
  const int lane = threadIdx.x & 31;
  const int idx  = blockIdx.x * 8 + wave;
  const int m0 = (idx & 3) << 5;                // node (c) base
  const int n0 = (idx >> 2) << 5;               // feature (f) base
  const int hs = lane >> 4;
  const int ln = lane & 15;

  const bf16* Ab = At  + (size_t)b * (RP * RP);
  const bf16* Bb = xsT + (size_t)b * (H * RP);

  v16bf a0 = load_a_frag(Ab, m0 + ln,      0, RP, hs);
  v16bf a1 = load_a_frag(Ab, m0 + 16 + ln, 0, RP, hs);
  v16bf b0 = load_b_frag(Bb, n0 + ln,      0, RP, hs);
  v16bf b1 = load_b_frag(Bb, n0 + 16 + ln, 0, RP, hs);
  v8f acc00 = {}, acc01 = {}, acc10 = {}, acc11 = {};

#pragma unroll
  for (int k0 = 0; k0 < RP; k0 += 32) {
    v16bf na0 = a0, na1 = a1, nb0 = b0, nb1 = b1;
    if (k0 + 32 < RP) {
      na0 = load_a_frag(Ab, m0 + ln,      k0 + 32, RP, hs);
      na1 = load_a_frag(Ab, m0 + 16 + ln, k0 + 32, RP, hs);
      nb0 = load_b_frag(Bb, n0 + ln,      k0 + 32, RP, hs);
      nb1 = load_b_frag(Bb, n0 + 16 + ln, k0 + 32, RP, hs);
    }
    acc00 = wmma_bf16(a0, b0, acc00);
    acc01 = wmma_bf16(a0, b1, acc01);
    acc10 = wmma_bf16(a1, b0, acc10);
    acc11 = wmma_bf16(a1, b1, acc11);
    a0 = na0; a1 = na1; b0 = nb0; b1 = nb1;
  }

  const float* dvb = dinv + (size_t)b * RP;
  bf16* hb = Hout + (size_t)b * (RP * H);
  const v8f* accs[2][2] = {{&acc00, &acc01}, {&acc10, &acc11}};
#pragma unroll
  for (int i = 0; i < 2; ++i) {
    const int c0 = m0 + 16 * i + 8 * hs;
    const float* dv = dvb + c0;
#pragma unroll
    for (int j = 0; j < 2; ++j) {
      const int f = n0 + 16 * j + ln;
      v8bf self = *(const v8bf*)(Bb + (size_t)f * RP + c0);  // xs[c0..c0+7][f]
      const float bv = bias[f];
      const v8f& acc = *accs[i][j];
      bf16* out = hb + (size_t)c0 * H + f;
#pragma unroll
      for (int e = 0; e < 8; ++e) {
        float v = dv[e] * (acc[e] + (float)self[e]) + bv;
        v = v > 0.f ? v : 0.2f * v;                          // leaky_relu(0.2)
        out[(size_t)e * H] = (bf16)v;
      }
    }
  }
}

// --------------------------- classifier ------------------------------------

__global__ __launch_bounds__(256) void classifier_kernel(
    const bf16* __restrict__ h2, const float* __restrict__ Wc,
    const float* __restrict__ bc, float* __restrict__ out) {
  const int b   = blockIdx.x;
  const int tid = threadIdx.x;
  const int N   = R * H;                       // 29696; pad rows (r>=116) excluded
  const bf16* hb = h2 + (size_t)b * (RP * H);  // row-major [r][f] -> flat r*H+f
  float a0 = 0.f, a1 = 0.f;
  for (int j = tid; j < N; j += 256) {
    float v = (float)hb[j];
    a0 += v * Wc[j];
    a1 += v * Wc[(size_t)N + j];
  }
  __shared__ float s0[256], s1[256];
  s0[tid] = a0; s1[tid] = a1;
  __syncthreads();
  for (int s = 128; s > 0; s >>= 1) {
    if (tid < s) { s0[tid] += s0[tid + s]; s1[tid] += s1[tid + s]; }
    __syncthreads();
  }
  if (tid == 0) {
    out[(size_t)b * 2 + 0] = s0[0] + bc[0];
    out[(size_t)b * 2 + 1] = s1[0] + bc[1];
  }
}

// --------------------------- launch ----------------------------------------

extern "C" void kernel_launch(void* const* d_in, const int* in_sizes, int n_in,
                              void* d_out, int out_size, void* d_ws, size_t ws_size,
                              hipStream_t stream) {
  (void)in_sizes; (void)n_in; (void)out_size; (void)ws_size;
  const float* m  = (const float*)d_in[0];
  const float* nf = (const float*)d_in[1];
  const float* W1 = (const float*)d_in[2];
  const float* b1 = (const float*)d_in[3];
  const float* W2 = (const float*)d_in[4];
  const float* b2 = (const float*)d_in[5];
  const float* Wc = (const float*)d_in[6];
  const float* bc = (const float*)d_in[7];
  float* out = (float*)d_out;

  char* ws = (char*)d_ws;
  size_t off = 0;
  auto carve = [&](size_t bytes) -> void* {
    void* p = ws + off;
    off += (bytes + 255) & ~(size_t)255;
    return p;
  };
  bf16*  At   = (bf16*)carve((size_t)BZ * RP * RP * 2);   // 16.8 MB  A^T bf16
  bf16*  Xb   = (bf16*)carve((size_t)BZ * RP * DP * 2);   // 16.8 MB  x bf16
  float* dinv = (float*)carve((size_t)BZ * RP * 4);       // 256 KB
  bf16*  W1t  = (bf16*)carve((size_t)H * DP * 2);         // 64 KB
  bf16*  W2t  = (bf16*)carve((size_t)H * H * 2);          // 128 KB
  bf16*  xsT  = (bf16*)carve((size_t)BZ * H * RP * 2);    // 33.6 MB (both layers)
  bf16*  Hbuf = (bf16*)carve((size_t)BZ * RP * H * 2);    // 33.6 MB (h1 then h2)

  prep_adj_kernel<<<BZ, RP, 0, stream>>>(m, At, dinv);
  prep_x_kernel  <<<BZ, DP, 0, stream>>>(nf, Xb);
  prep_w1_kernel <<<H,  DP, 0, stream>>>(W1, W1t);
  prep_w2_kernel <<<H,  H,  0, stream>>>(W2, W2t);

  dim3 g(4, BZ);  // 32 wave-tiles (32x32) per batch / 8 waves per block
  // Layer 1
  gemm_xs_kernel<DP><<<g, 256, 0, stream>>>(Xb, W1t, dinv, xsT);
  gemm_agg_kernel   <<<g, 256, 0, stream>>>(At, xsT, dinv, b1, Hbuf);
  // Layer 2 (Hbuf consumed by gemm_xs before gemm_agg overwrites it)
  gemm_xs_kernel<H> <<<g, 256, 0, stream>>>(Hbuf, W2t, dinv, xsT);
  gemm_agg_kernel   <<<g, 256, 0, stream>>>(At, xsT, dinv, b2, Hbuf);
  // Classifier
  classifier_kernel<<<BZ, 256, 0, stream>>>(Hbuf, Wc, bc, out);
}
